// EncoderLayer_27178553049805
// MI455X (gfx1250) — matrix operand order repaired and checked by reading, not compiled, not run on hardware
//
#include <hip/hip_runtime.h>
#include <hip/hip_bf16.h>

// ---------------------------------------------------------------------------
// Transformer encoder layer for MI455X (gfx1250, wave32, WMMA f16->f32).
// B=2 S=2048 D=768 H=12 HD=64 DFF=3072.
// Matmuls: v_wmma_f32_16x16x32_f16.  Tile staging: global_load_async_to_lds
// (ASYNCcnt) with double-buffered LDS pipelines.
// ---------------------------------------------------------------------------

typedef __attribute__((ext_vector_type(16))) _Float16 v16h;
typedef __attribute__((ext_vector_type(8)))  float    v8f;

#define B_   2
#define S_   2048
#define D_   768
#define H_   12
#define HD_  64
#define DFF_ 3072
#define MS_  (B_ * S_)      // 4096 rows

union HV { uint4 u[2]; v16h h; };

__device__ __forceinline__ v16h ld_frag(const _Float16* p0, const _Float16* p1) {
    HV r;
    r.u[0] = *(const uint4*)p0;
    r.u[1] = *(const uint4*)p1;
    return r.h;
}

__device__ __forceinline__ v8f wmma32(v16h a, v16h b, v8f c) {
    return __builtin_amdgcn_wmma_f32_16x16x32_f16(
        /*neg_a=*/false, a, /*neg_b=*/false, b,
        /*c_mod=*/(short)0, c, /*reuse_a=*/false, /*reuse_b=*/false);
}

// ---- CDNA5 async global->LDS copy (16B per lane, tracked by ASYNCcnt) ----
__device__ __forceinline__ void cp_async_b128(const void* g, void* l) {
    const unsigned int lds_off = (unsigned int)(unsigned long long)l; // generic LDS addr truncates to LDS offset (ISA 10.2)
    const unsigned long long ga = (unsigned long long)g;
    asm volatile("global_load_async_to_lds_b128 %0, %1, off"
                 :: "v"(lds_off), "v"(ga) : "memory");
}
__device__ __forceinline__ void cp_async_wait() {
    asm volatile("s_wait_asynccnt 0" ::: "memory");
}

// ------------------------------ converters --------------------------------
__global__ void cvt_h_kernel(const float* __restrict__ in, _Float16* __restrict__ out, int n) {
    int i = blockIdx.x * blockDim.x + threadIdx.x;
    if (i < n) out[i] = (_Float16)in[i];
}

// in [K,N] f32 row-major -> out [N,K] f16 (transposed for B-fragment reads)
__global__ void cvt_tr_kernel(const float* __restrict__ in, _Float16* __restrict__ out, int K, int N) {
    int i = blockIdx.x * blockDim.x + threadIdx.x;
    if (i < K * N) {
        int n = i / K, k = i % K;
        out[i] = (_Float16)in[(size_t)k * N + n];
    }
}

// ------------------------------- GEMM -------------------------------------
// Y[M,N] = A[M,K] @ Bt[N,K]^T + bias, optional relu; writes f32 and/or f16.
// Block tile 128x128x32, 8 waves (4 along M x 2 along N), wave tile 32x64.
// Double-buffered LDS, async copy pipeline.
#define BM 128
#define BN 128
#define BK 32

__global__ __launch_bounds__(256) void gemm_f16_kernel(
    const _Float16* __restrict__ A, const _Float16* __restrict__ Bt,
    const float* __restrict__ bias, float* __restrict__ outF,
    _Float16* __restrict__ outH, int M, int N, int K, int relu)
{
    __shared__ __align__(16) _Float16 As[2][BM][BK];
    __shared__ __align__(16) _Float16 Bs[2][BM][BK];

    const int tid  = threadIdx.x;
    const int wid  = tid >> 5;
    const int lane = tid & 31;
    const int hl   = lane >> 4;     // half-wave (0/1)
    const int l15  = lane & 15;

    const int m0 = blockIdx.x * BM;
    const int n0 = blockIdx.y * BN;
    const int wm = (wid & 3) * 32;  // wave row offset in tile
    const int wn = (wid >> 2) * 64; // wave col offset in tile

    v8f acc[2][4];
    for (int i = 0; i < 2; i++)
        for (int j = 0; j < 4; j++)
            acc[i][j] = {};

    const int lr = tid >> 2;         // 0..63
    const int lc = (tid & 3) * 8;    // 0,8,16,24

    auto issue = [&](int k0, int buf) {
        cp_async_b128(&A [(size_t)(m0 + lr)      * K + k0 + lc], &As[buf][lr][lc]);
        cp_async_b128(&A [(size_t)(m0 + lr + 64) * K + k0 + lc], &As[buf][lr + 64][lc]);
        cp_async_b128(&Bt[(size_t)(n0 + lr)      * K + k0 + lc], &Bs[buf][lr][lc]);
        cp_async_b128(&Bt[(size_t)(n0 + lr + 64) * K + k0 + lc], &Bs[buf][lr + 64][lc]);
    };

    issue(0, 0);
    const int nk = K / BK;
    for (int kc = 0; kc < nk; kc++) {
        cp_async_wait();            // this wave's tile kc is in LDS
        __syncthreads();            // everyone's tile kc is in LDS
        if (kc + 1 < nk) issue((kc + 1) * BK, (kc + 1) & 1);

        const int cur = kc & 1;
        v16h af[2], bf[4];
        for (int i = 0; i < 2; i++) {
            const _Float16* p = &As[cur][wm + i * 16 + l15][hl * 8];
            af[i] = ld_frag(p, p + 16);          // K = hl*8+[0..7], 16+hl*8+[0..7]
        }
        for (int j = 0; j < 4; j++) {
            const _Float16* p = &Bs[cur][wn + j * 16 + l15][hl * 8];
            bf[j] = ld_frag(p, p + 16);          // K = hl*8+[0..7], 16+hl*8+[0..7]
        }
        for (int i = 0; i < 2; i++)
            for (int j = 0; j < 4; j++)
                acc[i][j] = wmma32(af[i], bf[j], acc[i][j]);
        __syncthreads();            // all reads of buf `cur` done before rewrite
    }

    // epilogue: C layout => lane = col, vgpr r = row (r + 8*hl)
    for (int j = 0; j < 4; j++) {
        const int n = n0 + wn + j * 16 + l15;
        const float bv = bias ? bias[n] : 0.0f;
        for (int i = 0; i < 2; i++) {
            for (int r = 0; r < 8; r++) {
                const int m = m0 + wm + i * 16 + r + hl * 8;
                float v = acc[i][j][r] + bv;
                if (relu) v = fmaxf(v, 0.0f);
                if (outF) outF[(size_t)m * N + n] = v;
                if (outH) outH[(size_t)m * N + n] = (_Float16)v;
            }
        }
    }
}

// --------------------------- flash attention ------------------------------
// grid = (S/128, B*H); 8 waves/block, each wave owns 16 q rows.
// Double-buffered 32-key K/V tiles (K via async copy, V transposed in-flight);
// online softmax; P staged through per-wave LDS to become the A-fragment.
__global__ __launch_bounds__(256) void attn_kernel(
    const _Float16* __restrict__ Qh, const _Float16* __restrict__ Kh,
    const _Float16* __restrict__ Vh, const int* __restrict__ mask,
    _Float16* __restrict__ Oh)
{
    __shared__ __align__(16) _Float16 Kt[2][32][HD_];   // [buf][key][hd]
    __shared__ __align__(16) _Float16 Vt[2][HD_][32];   // [buf][hd][key] (transposed)
    __shared__ __align__(16) _Float16 Pw[8][16][32];    // per-wave P staging

    const int tid  = threadIdx.x;
    const int wid  = tid >> 5;
    const int lane = tid & 31;
    const int hl   = lane >> 4;
    const int l15  = lane & 15;

    const int b  = blockIdx.y / H_;
    const int h  = blockIdx.y % H_;
    const int q0 = blockIdx.x * 128 + wid * 16;
    const float scale = 0.125f;                         // 1/sqrt(64)

    // Q A-fragments (rows q0+l15), two 32-wide K chunks over HD=64
    v16h qa[2];
    {
        const size_t rb = (size_t)(b * S_ + q0 + l15) * D_ + h * HD_;
        for (int c = 0; c < 2; c++) {
            const _Float16* p = &Qh[rb + c * 32 + hl * 8];
            qa[c] = ld_frag(p, p + 16);
        }
    }

    v8f acc[4];
    for (int c = 0; c < 4; c++) acc[c] = {};
    float mrow[8], lrow[8];
    for (int r = 0; r < 8; r++) { mrow[r] = -1e30f; lrow[r] = 0.0f; }

    const int lkey = tid >> 3;            // 0..31
    const int lhd  = (tid & 7) * 8;       // 0..56

    auto issue = [&](int k0, int buf) {
        const size_t gb = (size_t)(b * S_ + k0 + lkey) * D_ + h * HD_ + lhd;
        cp_async_b128(&Kh[gb], &Kt[buf][lkey][lhd]);    // async K tile
        uint4 vv = *(const uint4*)&Vh[gb];              // V tile, transposed store
        const _Float16* vh = (const _Float16*)&vv;
        for (int j = 0; j < 8; j++) Vt[buf][lhd + j][lkey] = vh[j];
    };

    issue(0, 0);
    const int nt = S_ / 32;
    for (int kt = 0; kt < nt; kt++) {
        cp_async_wait();
        __syncthreads();
        if (kt + 1 < nt) issue((kt + 1) * 32, (kt + 1) & 1);
        const int cur = kt & 1;
        const int k0  = kt * 32;

        // energy tiles: e0 = keys[0..15], e1 = keys[16..31]
        v8f e0 = {}, e1 = {};
        for (int c = 0; c < 2; c++) {
            const _Float16* p0 = &Kt[cur][l15][c * 32 + hl * 8];
            const _Float16* p1 = &Kt[cur][16 + l15][c * 32 + hl * 8];
            v16h kb0 = ld_frag(p0, p0 + 16);
            v16h kb1 = ld_frag(p1, p1 + 16);
            e0 = wmma32(qa[c], kb0, e0);
            e1 = wmma32(qa[c], kb1, e1);
        }

        const int mk0 = mask[b * S_ + k0 + l15];
        const int mk1 = mask[b * S_ + k0 + 16 + l15];

        for (int r = 0; r < 8; r++) {
            float a0 = mk0 ? e0[r] * scale : -1e30f;
            float a1 = mk1 ? e1[r] * scale : -1e30f;
            // row max across the 16 lanes of this half-wave
            float mx = fmaxf(a0, a1);
            for (int s = 8; s >= 1; s >>= 1) mx = fmaxf(mx, __shfl_xor(mx, s));
            const float mn   = fmaxf(mrow[r], mx);
            const float corr = __expf(mrow[r] - mn);
            const float pe0  = __expf(a0 - mn);
            const float pe1  = __expf(a1 - mn);
            float sum = pe0 + pe1;
            for (int s = 8; s >= 1; s >>= 1) sum += __shfl_xor(sum, s);
            lrow[r] = lrow[r] * corr + sum;
            mrow[r] = mn;
            for (int c = 0; c < 4; c++) acc[c][r] = acc[c][r] * corr;
            const int m = r + hl * 8;
            Pw[wid][m][l15]      = (_Float16)pe0;
            Pw[wid][m][16 + l15] = (_Float16)pe1;
        }

        // out += P(16x32) @ V(32xHD)
        const _Float16* pp = &Pw[wid][l15][hl * 8];
        v16h pa = ld_frag(pp, pp + 16);
        for (int c = 0; c < 4; c++) {
            const _Float16* vp = &Vt[cur][c * 16 + l15][hl * 8];
            v16h vb = ld_frag(vp, vp + 16);
            acc[c] = wmma32(pa, vb, acc[c]);
        }
        __syncthreads();
    }

    // normalize and store f16 output (row-major [b*S+s][h*HD+hd])
    for (int c = 0; c < 4; c++) {
        for (int r = 0; r < 8; r++) {
            const int m = r + hl * 8;
            const float v = acc[c][r] / lrow[r];
            Oh[(size_t)(b * S_ + q0 + m) * D_ + h * HD_ + c * 16 + l15] = (_Float16)v;
        }
    }
}

// ------------------------ residual + layernorm ----------------------------
__global__ __launch_bounds__(256) void add_ln_kernel(
    const float* __restrict__ res, const float* __restrict__ add,
    const float* __restrict__ g, const float* __restrict__ be,
    float* __restrict__ outF, _Float16* __restrict__ outH)
{
    __shared__ float red[256];
    const int row = blockIdx.x, t = threadIdx.x;
    const size_t base = (size_t)row * D_;

    float x0 = res[base + t]       + add[base + t];
    float x1 = res[base + t + 256] + add[base + t + 256];
    float x2 = res[base + t + 512] + add[base + t + 512];

    red[t] = x0 + x1 + x2;
    __syncthreads();
    for (int o = 128; o > 0; o >>= 1) { if (t < o) red[t] += red[t + o]; __syncthreads(); }
    const float mu = red[0] * (1.0f / (float)D_);
    __syncthreads();

    const float d0 = x0 - mu, d1 = x1 - mu, d2 = x2 - mu;
    red[t] = d0 * d0 + d1 * d1 + d2 * d2;
    __syncthreads();
    for (int o = 128; o > 0; o >>= 1) { if (t < o) red[t] += red[t + o]; __syncthreads(); }
    const float rs = rsqrtf(red[0] * (1.0f / (float)D_) + 1e-5f);

    float y0 = d0 * rs * g[t]       + be[t];
    float y1 = d1 * rs * g[t + 256] + be[t + 256];
    float y2 = d2 * rs * g[t + 512] + be[t + 512];
    outF[base + t]       = y0;
    outF[base + t + 256] = y1;
    outF[base + t + 512] = y2;
    if (outH) {
        outH[base + t]       = (_Float16)y0;
        outH[base + t + 256] = (_Float16)y1;
        outH[base + t + 512] = (_Float16)y2;
    }
}

// ------------------------------ launcher ----------------------------------
extern "C" void kernel_launch(void* const* d_in, const int* in_sizes, int n_in,
                              void* d_out, int out_size, void* d_ws, size_t ws_size,
                              hipStream_t stream) {
    const float* src  = (const float*)d_in[0];
    const int*   mask = (const int*)  d_in[1];
    const float* Wq = (const float*)d_in[2];  const float* bq = (const float*)d_in[3];
    const float* Wk = (const float*)d_in[4];  const float* bk = (const float*)d_in[5];
    const float* Wv = (const float*)d_in[6];  const float* bv = (const float*)d_in[7];
    const float* Wo = (const float*)d_in[8];  const float* bo = (const float*)d_in[9];
    const float* W1 = (const float*)d_in[10]; const float* b1 = (const float*)d_in[11];
    const float* W2 = (const float*)d_in[12]; const float* b2 = (const float*)d_in[13];
    const float* g1 = (const float*)d_in[14]; const float* be1 = (const float*)d_in[15];
    const float* g2 = (const float*)d_in[16]; const float* be2 = (const float*)d_in[17];
    float* out = (float*)d_out;

    // workspace carve-up
    char* ws = (char*)d_ws;
    size_t off = 0;
    auto alloc = [&](size_t bytes) -> void* {
        void* p = ws + off;
        off = (off + bytes + 255) & ~(size_t)255;
        return p;
    };
    _Float16* srcH = (_Float16*)alloc((size_t)MS_ * D_ * 2);
    _Float16* WqT  = (_Float16*)alloc((size_t)D_ * D_ * 2);
    _Float16* WkT  = (_Float16*)alloc((size_t)D_ * D_ * 2);
    _Float16* WvT  = (_Float16*)alloc((size_t)D_ * D_ * 2);
    _Float16* WoT  = (_Float16*)alloc((size_t)D_ * D_ * 2);
    _Float16* W1T  = (_Float16*)alloc((size_t)D_ * DFF_ * 2);
    _Float16* W2T  = (_Float16*)alloc((size_t)D_ * DFF_ * 2);
    _Float16* QhB  = (_Float16*)alloc((size_t)MS_ * D_ * 2);
    _Float16* KhB  = (_Float16*)alloc((size_t)MS_ * D_ * 2);
    _Float16* VhB  = (_Float16*)alloc((size_t)MS_ * D_ * 2);
    _Float16* OhB  = (_Float16*)alloc((size_t)MS_ * D_ * 2);
    float*    attnF= (float*)   alloc((size_t)MS_ * D_ * 4);
    float*    xF   = (float*)   alloc((size_t)MS_ * D_ * 4);
    _Float16* xH   = (_Float16*)alloc((size_t)MS_ * D_ * 2);
    _Float16* ffH  = (_Float16*)alloc((size_t)MS_ * DFF_ * 2);
    float*    ff2F = (float*)   alloc((size_t)MS_ * D_ * 4);
    (void)ws_size; (void)n_in; (void)in_sizes; (void)out_size;

    // 1) f16 conversions (weights transposed for B-fragment-friendly layout)
    cvt_h_kernel<<<(MS_ * D_ + 255) / 256, 256, 0, stream>>>(src, srcH, MS_ * D_);
    cvt_tr_kernel<<<(D_ * D_ + 255) / 256, 256, 0, stream>>>(Wq, WqT, D_, D_);
    cvt_tr_kernel<<<(D_ * D_ + 255) / 256, 256, 0, stream>>>(Wk, WkT, D_, D_);
    cvt_tr_kernel<<<(D_ * D_ + 255) / 256, 256, 0, stream>>>(Wv, WvT, D_, D_);
    cvt_tr_kernel<<<(D_ * D_ + 255) / 256, 256, 0, stream>>>(Wo, WoT, D_, D_);
    cvt_tr_kernel<<<(D_ * DFF_ + 255) / 256, 256, 0, stream>>>(W1, W1T, D_, DFF_);
    cvt_tr_kernel<<<(D_ * DFF_ + 255) / 256, 256, 0, stream>>>(W2, W2T, DFF_, D_);

    // 2) Q/K/V projections
    dim3 gD(MS_ / BM, D_ / BN);
    gemm_f16_kernel<<<gD, 256, 0, stream>>>(srcH, WqT, bq, nullptr, QhB, MS_, D_, D_, 0);
    gemm_f16_kernel<<<gD, 256, 0, stream>>>(srcH, WkT, bk, nullptr, KhB, MS_, D_, D_, 0);
    gemm_f16_kernel<<<gD, 256, 0, stream>>>(srcH, WvT, bv, nullptr, VhB, MS_, D_, D_, 0);

    // 3) flash attention
    dim3 gA(S_ / 128, B_ * H_);
    attn_kernel<<<gA, 256, 0, stream>>>(QhB, KhB, VhB, mask, OhB);

    // 4) output projection + residual LN1
    gemm_f16_kernel<<<gD, 256, 0, stream>>>(OhB, WoT, bo, attnF, nullptr, MS_, D_, D_, 0);
    add_ln_kernel<<<MS_, 256, 0, stream>>>(src, attnF, g1, be1, xF, xH);

    // 5) FFN
    dim3 gF1(MS_ / BM, DFF_ / BN);
    gemm_f16_kernel<<<gF1, 256, 0, stream>>>(xH, W1T, b1, nullptr, ffH, MS_, DFF_, D_, 1);
    gemm_f16_kernel<<<gD, 256, 0, stream>>>(ffH, W2T, b2, ff2F, nullptr, MS_, D_, DFF_, 0);

    // 6) residual LN2 -> final output (f32)
    add_ln_kernel<<<MS_, 256, 0, stream>>>(xF, ff2F, g2, be2, out, nullptr);
}
